// SA_Layer_Linear_80882824119000
// MI455X (gfx1250) — matrix operand order repaired and checked by reading, not compiled, not run on hardware
//
#include <hip/hip_runtime.h>

typedef unsigned short u16;
typedef __attribute__((ext_vector_type(16))) __bf16 v16bf;
typedef __attribute__((ext_vector_type(8)))  float  v8f;

#define BM 128
#define BN 256
#define BK 32
#define LDSS 40   // padded LDS row stride in u16 units (80 bytes, 16B-aligned)

__device__ __forceinline__ u16 f2bf(float f) {
  union { float f; unsigned int u; } x; x.f = f;
  unsigned int u = x.u;
  unsigned int r = (u + 0x7FFFu + ((u >> 16) & 1u)) >> 16;
  return (u16)r;
}
__device__ __forceinline__ float bf2f(u16 h) {
  union { unsigned int u; float f; } x; x.u = ((unsigned int)h) << 16;
  return x.f;
}

union FragU { uint4 q[2]; v16bf v; };
struct Acc16 { v8f a[4][4]; };

// ---------------------------------------------------------------------------
// Shared WMMA GEMM core: out[m0:m0+128, j0:j0+256] = A[M,K] * Bt[Ncols,K]^T
// 256 threads = 8 waves (2x4). Wave tile 64x64 -> 4x4 fragments of 16x16:
// 16 ds_load_b128 per K-step feeding 16 v_wmma (ds:wmma = 1.0).
// A fragment (16x32 bf16): lane m=l16, half selects K {0-7,16-23} / {8-15,24-31}.
// B fragment (32x16 bf16): lane n=l16, half selects K {0-7,16-23} / {8-15,24-31}.
// ---------------------------------------------------------------------------
__device__ __forceinline__ void gemm_core(const u16* __restrict__ A,
                                          const u16* __restrict__ Bt,
                                          int K, int m0, int j0,
                                          u16* As, u16* Bs, Acc16& acc) {
  const int tid  = threadIdx.x;
  const int lane = tid & 31;
  const int wave = tid >> 5;
  const int wm   = wave >> 2;     // 0..1
  const int wn   = wave & 3;      // 0..3
  const int half = lane >> 4;     // 0..1
  const int l16  = lane & 15;
  v8f zero = {0.f,0.f,0.f,0.f,0.f,0.f,0.f,0.f};
#pragma unroll
  for (int i = 0; i < 4; ++i)
#pragma unroll
    for (int j = 0; j < 4; ++j) acc.a[i][j] = zero;

  const int lrow  = tid >> 1;         // 0..127  (A staging: 2 threads/row)
  const int lkoff = (tid & 1) * 16;   // 0 or 16

  for (int k0 = 0; k0 < K; k0 += BK) {
    const u16* ag = A  + (size_t)(m0 + lrow) * K + (size_t)(k0 + lkoff);
    uint4 a0 = *(const uint4*)(ag);
    uint4 a1 = *(const uint4*)(ag + 8);
    const u16* bg = Bt + (size_t)(j0 + tid) * K + (size_t)k0;  // B staging: 1 thread/row
    uint4 b0 = *(const uint4*)(bg);
    uint4 b1 = *(const uint4*)(bg + 8);
    uint4 b2 = *(const uint4*)(bg + 16);
    uint4 b3 = *(const uint4*)(bg + 24);
    __syncthreads();
    *(uint4*)(As + lrow * LDSS + lkoff)     = a0;
    *(uint4*)(As + lrow * LDSS + lkoff + 8) = a1;
    *(uint4*)(Bs + tid * LDSS)      = b0;
    *(uint4*)(Bs + tid * LDSS + 8)  = b1;
    *(uint4*)(Bs + tid * LDSS + 16) = b2;
    *(uint4*)(Bs + tid * LDSS + 24) = b3;
    __syncthreads();

    FragU bf[4];
#pragma unroll
    for (int j = 0; j < 4; ++j) {
      const u16* rp = Bs + (wn * 64 + j * 16 + l16) * LDSS;
      bf[j].q[0] = *(const uint4*)(rp + half * 8);
      bf[j].q[1] = *(const uint4*)(rp + 16 + half * 8);
    }
#pragma unroll
    for (int i = 0; i < 4; ++i) {
      FragU af;
      const u16* rp = As + (wm * 64 + i * 16 + l16) * LDSS;
      af.q[0] = *(const uint4*)(rp + half * 8);
      af.q[1] = *(const uint4*)(rp + 16 + half * 8);
#pragma unroll
      for (int j = 0; j < 4; ++j)
        acc.a[i][j] = __builtin_amdgcn_wmma_f32_16x16x32_bf16(
            false, af.v, false, bf[j].v, (short)0, acc.a[i][j], false, false);
    }
  }
}

#define EPI_VARS \
  const int tid = threadIdx.x; const int lane = tid & 31; const int wave = tid >> 5; \
  const int wm = wave >> 2; const int wn = wave & 3; \
  const int half = lane >> 4; const int l16 = lane & 15;

// q/k projection: val = relu(acc)+1 ; store bf16 into qc[b][m][n] and qt[b][n][m]
__global__ __launch_bounds__(256)
void gemm_qk_kernel(const u16* __restrict__ W, const u16* __restrict__ xt,
                    u16* __restrict__ qc, u16* __restrict__ qt, int C, int N) {
  __shared__ u16 As[BM * LDSS]; __shared__ u16 Bs[BN * LDSS];
  const int b = blockIdx.z;
  const int m0 = blockIdx.y * BM, j0 = blockIdx.x * BN;
  Acc16 acc;
  gemm_core(W, xt + (size_t)b * N * C, C, m0, j0, As, Bs, acc);
  EPI_VARS
#pragma unroll
  for (int i = 0; i < 4; ++i)
#pragma unroll
    for (int j = 0; j < 4; ++j)
#pragma unroll
      for (int r = 0; r < 8; ++r) {
        int m = m0 + wm * 64 + i * 16 + half * 8 + r;
        int n = j0 + wn * 64 + j * 16 + l16;
        u16 h = f2bf(fmaxf(acc.a[i][j][r], 0.f) + 1.f);
        qc[((size_t)b * C + m) * N + n] = h;
        qt[((size_t)b * N + n) * C + m] = h;
      }
}

// v projection: bf16 row-major store vc[b][m][n]
__global__ __launch_bounds__(256)
void gemm_v_kernel(const u16* __restrict__ W, const u16* __restrict__ xt,
                   u16* __restrict__ vc, int C, int N) {
  __shared__ u16 As[BM * LDSS]; __shared__ u16 Bs[BN * LDSS];
  const int b = blockIdx.z;
  const int m0 = blockIdx.y * BM, j0 = blockIdx.x * BN;
  Acc16 acc;
  gemm_core(W, xt + (size_t)b * N * C, C, m0, j0, As, Bs, acc);
  EPI_VARS
#pragma unroll
  for (int i = 0; i < 4; ++i)
#pragma unroll
    for (int j = 0; j < 4; ++j)
#pragma unroll
      for (int r = 0; r < 8; ++r) {
        int m = m0 + wm * 64 + i * 16 + half * 8 + r;
        int n = j0 + wn * 64 + j * 16 + l16;
        vc[((size_t)b * C + m) * N + n] = f2bf(acc.a[i][j][r]);
      }
}

// phi projection: f32 row-major store p[b][m][n]
__global__ __launch_bounds__(256)
void gemm_phi_kernel(const u16* __restrict__ W, const u16* __restrict__ xt,
                     float* __restrict__ p, int C, int N) {
  __shared__ u16 As[BM * LDSS]; __shared__ u16 Bs[BN * LDSS];
  const int b = blockIdx.z;
  const int m0 = blockIdx.y * BM, j0 = blockIdx.x * BN;
  Acc16 acc;
  gemm_core(W, xt + (size_t)b * N * C, C, m0, j0, As, Bs, acc);
  EPI_VARS
#pragma unroll
  for (int i = 0; i < 4; ++i)
#pragma unroll
    for (int j = 0; j < 4; ++j)
#pragma unroll
      for (int r = 0; r < 8; ++r) {
        int m = m0 + wm * 64 + i * 16 + half * 8 + r;
        int n = j0 + wn * 64 + j * 16 + l16;
        p[((size_t)b * C + m) * N + n] = acc.a[i][j][r];
      }
}

// kv[c,d] = (1/sqrt(C)) * sum_n khat[c,n] v[d,n]; store transposed bf16 kvt[b][d][c]
__global__ __launch_bounds__(256)
void gemm_kv_kernel(const u16* __restrict__ khat, const u16* __restrict__ vc,
                    u16* __restrict__ kvt, int C, int N, float scale) {
  __shared__ u16 As[BM * LDSS]; __shared__ u16 Bs[BN * LDSS];
  const int b = blockIdx.z;
  const int m0 = blockIdx.y * BM, j0 = blockIdx.x * BN;
  Acc16 acc;
  gemm_core(khat + (size_t)b * C * N, vc + (size_t)b * C * N, N, m0, j0, As, Bs, acc);
  EPI_VARS
#pragma unroll
  for (int i = 0; i < 4; ++i)
#pragma unroll
    for (int j = 0; j < 4; ++j)
#pragma unroll
      for (int r = 0; r < 8; ++r) {
        int m = m0 + wm * 64 + i * 16 + half * 8 + r;    // c
        int d = j0 + wn * 64 + j * 16 + l16;             // d
        kvt[((size_t)b * C + d) * C + m] = f2bf(acc.a[i][j][r] * scale);
      }
}

// y[d,n] = (kv^T q)[d,n] * zinv[n]; u^T[n][d] = x[d][n] + y*p[d][n]
__global__ __launch_bounds__(256)
void gemm_y_kernel(const u16* __restrict__ kvt, const u16* __restrict__ qt,
                   const float* __restrict__ zinv, const float* __restrict__ x,
                   const float* __restrict__ p, float* __restrict__ uT,
                   int C, int N) {
  __shared__ u16 As[BM * LDSS]; __shared__ u16 Bs[BN * LDSS];
  const int b = blockIdx.z;
  const int m0 = blockIdx.y * BM, j0 = blockIdx.x * BN;
  Acc16 acc;
  gemm_core(kvt + (size_t)b * C * C, qt + (size_t)b * N * C, C, m0, j0, As, Bs, acc);
  EPI_VARS
#pragma unroll
  for (int i = 0; i < 4; ++i)
#pragma unroll
    for (int j = 0; j < 4; ++j)
#pragma unroll
      for (int r = 0; r < 8; ++r) {
        int m = m0 + wm * 64 + i * 16 + half * 8 + r;    // d (channel)
        int n = j0 + wn * 64 + j * 16 + l16;             // point
        float y = acc.a[i][j][r] * zinv[(size_t)b * N + n];
        size_t ri = ((size_t)b * C + m) * N + n;
        uT[((size_t)b * N + n) * C + m] = x[ri] + y * p[ri];
      }
}

// FFN1: g^T[j][f] = relu(w1 h^T + bb1)  (rows j = flattened B*N tokens)
__global__ __launch_bounds__(256)
void gemm_ffn1_kernel(const u16* __restrict__ w1b, const u16* __restrict__ hb,
                      const float* __restrict__ bb1, u16* __restrict__ gT,
                      int C, int F) {
  __shared__ u16 As[BM * LDSS]; __shared__ u16 Bs[BN * LDSS];
  const int m0 = blockIdx.y * BM, j0 = blockIdx.x * BN;
  Acc16 acc;
  gemm_core(w1b, hb, C, m0, j0, As, Bs, acc);
  EPI_VARS
#pragma unroll
  for (int i = 0; i < 4; ++i)
#pragma unroll
    for (int j = 0; j < 4; ++j)
#pragma unroll
      for (int r = 0; r < 8; ++r) {
        int m = m0 + wm * 64 + i * 16 + half * 8 + r;    // f
        int n = j0 + wn * 64 + j * 16 + l16;             // token
        gT[(size_t)n * F + m] = f2bf(fmaxf(acc.a[i][j][r] + bb1[m], 0.f));
      }
}

// FFN2: t[j][c] = w2 g^T + bb2 + h_f32[j][c]
__global__ __launch_bounds__(256)
void gemm_ffn2_kernel(const u16* __restrict__ w2b, const u16* __restrict__ gT,
                      const float* __restrict__ bb2, const float* __restrict__ hf,
                      float* __restrict__ t, int F, int C) {
  __shared__ u16 As[BM * LDSS]; __shared__ u16 Bs[BN * LDSS];
  const int m0 = blockIdx.y * BM, j0 = blockIdx.x * BN;
  Acc16 acc;
  gemm_core(w2b, gT, F, m0, j0, As, Bs, acc);
  EPI_VARS
#pragma unroll
  for (int i = 0; i < 4; ++i)
#pragma unroll
    for (int j = 0; j < 4; ++j)
#pragma unroll
      for (int r = 0; r < 8; ++r) {
        int m = m0 + wm * 64 + i * 16 + half * 8 + r;    // c
        int n = j0 + wn * 64 + j * 16 + l16;             // token
        size_t idx = (size_t)n * C + m;
        t[idx] = acc.a[i][j][r] + bb2[m] + hf[idx];
      }
}

// ---------------------------------------------------------------------------
// Small f32 helper kernels
// ---------------------------------------------------------------------------
__global__ void cvt_f32_bf16_kernel(const float* __restrict__ in,
                                    u16* __restrict__ out, size_t n) {
  size_t i = (size_t)blockIdx.x * blockDim.x + threadIdx.x;
  if (i < n) out[i] = f2bf(in[i]);
}

// x[b][c][n] f32 -> xt[b][n][c] bf16 (32x32 LDS tile)
__global__ __launch_bounds__(256)
void transpose_cvt_kernel(const float* __restrict__ x, u16* __restrict__ xt,
                          int C, int N) {
  __shared__ float tile[32][33];
  const int b = blockIdx.z;
  const int n0 = blockIdx.x * 32, c0 = blockIdx.y * 32;
  const int tx = threadIdx.x & 31, ty = threadIdx.x >> 5;
#pragma unroll
  for (int i = 0; i < 32; i += 8)
    tile[ty + i][tx] = x[((size_t)b * C + c0 + ty + i) * N + n0 + tx];
  __syncthreads();
#pragma unroll
  for (int i = 0; i < 32; i += 8)
    xt[((size_t)b * N + n0 + ty + i) * C + c0 + tx] = f2bf(tile[tx][ty + i]);
}

// out[row] = scale * sum_n in_bf16[row][n]   (row = b*C + c, wave per row)
__global__ __launch_bounds__(256)
void rowsum_bf16_kernel(const u16* __restrict__ in, float* __restrict__ out,
                        int N, float scale) {
  const int row = blockIdx.x * 8 + (threadIdx.x >> 5);
  const int lane = threadIdx.x & 31;
  const u16* p = in + (size_t)row * N;
  float s = 0.f;
  for (int n = lane; n < N; n += 32) s += bf2f(p[n]);
#pragma unroll
  for (int o = 16; o; o >>= 1) s += __shfl_xor(s, o, 32);
  if (lane == 0) out[row] = s * scale;
}

// out[row=b*N+n] = f( sum_c qt[row][c] * vec[b][c] ); recip -> 1/(x+1e-6)
__global__ __launch_bounds__(256)
void rowdot_kernel(const u16* __restrict__ qt, const float* __restrict__ vec,
                   float* __restrict__ out, int C, int N, int recip) {
  const int row = blockIdx.x * 8 + (threadIdx.x >> 5);
  const int lane = threadIdx.x & 31;
  const int b = row / N;
  const u16* p = qt + (size_t)row * C;
  const float* v = vec + (size_t)b * C;
  float s = 0.f;
  for (int c = lane; c < C; c += 32) s += bf2f(p[c]) * v[c];
#pragma unroll
  for (int o = 16; o; o >>= 1) s += __shfl_xor(s, o, 32);
  if (lane == 0) out[row] = recip ? 1.f / (s + 1e-6f) : s;
}

// alpha[b][n] = softmax_n(s)[n] * N   (one block per batch)
__global__ __launch_bounds__(1024)
void softmax_scale_kernel(const float* __restrict__ s, float* __restrict__ alpha,
                          int N) {
  const int b = blockIdx.x;
  const float* sp = s + (size_t)b * N;
  float* ap = alpha + (size_t)b * N;
  const int tid = threadIdx.x, lane = tid & 31, wave = tid >> 5;
  __shared__ float red[32];
  __shared__ float s_max, s_sum;
  float m = -3.4e38f;
  for (int n = tid; n < N; n += 1024) m = fmaxf(m, sp[n]);
#pragma unroll
  for (int o = 16; o; o >>= 1) m = fmaxf(m, __shfl_xor(m, o, 32));
  if (lane == 0) red[wave] = m;
  __syncthreads();
  if (wave == 0) {
    float t = red[lane];
#pragma unroll
    for (int o = 16; o; o >>= 1) t = fmaxf(t, __shfl_xor(t, o, 32));
    if (lane == 0) s_max = t;
  }
  __syncthreads();
  float e = 0.f;
  for (int n = tid; n < N; n += 1024) e += __expf(sp[n] - s_max);
#pragma unroll
  for (int o = 16; o; o >>= 1) e += __shfl_xor(e, o, 32);
  if (lane == 0) red[wave] = e;
  __syncthreads();
  if (wave == 0) {
    float t = red[lane];
#pragma unroll
    for (int o = 16; o; o >>= 1) t += __shfl_xor(t, o, 32);
    if (lane == 0) s_sum = t;
  }
  __syncthreads();
  const float inv = (float)N / s_sum;
  for (int n = tid; n < N; n += 1024) ap[n] = __expf(sp[n] - s_max) * inv;
}

// khat[b][c][n] = bf16( qc[b][c][n] * alpha[b][n] )
__global__ void khat_kernel(const u16* __restrict__ qc, const float* __restrict__ alpha,
                            u16* __restrict__ khat, int C, int N, size_t total) {
  size_t i = (size_t)blockIdx.x * blockDim.x + threadIdx.x;
  if (i >= total) return;
  int n = (int)(i % (size_t)N);
  size_t b = i / ((size_t)C * N);
  khat[i] = f2bf(bf2f(qc[i]) * alpha[b * N + n]);
}

// LayerNorm over C (wave per row): writes bf16 + f32 copies
__global__ __launch_bounds__(256)
void ln1_kernel(const float* __restrict__ in, const float* __restrict__ g,
                const float* __restrict__ bta, u16* __restrict__ out_bf,
                float* __restrict__ out_f, int C) {
  const int row = blockIdx.x * 8 + (threadIdx.x >> 5);
  const int lane = threadIdx.x & 31;
  const float* p = in + (size_t)row * C;
  float s = 0.f, ss = 0.f;
  for (int c = lane; c < C; c += 32) { float v = p[c]; s += v; ss += v * v; }
#pragma unroll
  for (int o = 16; o; o >>= 1) { s += __shfl_xor(s, o, 32); ss += __shfl_xor(ss, o, 32); }
  const float mu = s / C;
  const float r = rsqrtf(ss / C - mu * mu + 1e-6f);
  for (int c = lane; c < C; c += 32) {
    float v = (p[c] - mu) * r * g[c] + bta[c];
    out_bf[(size_t)row * C + c] = f2bf(v);
    out_f[(size_t)row * C + c] = v;
  }
}

// Final LN + relu + transpose to out[b][c][n]
__global__ __launch_bounds__(256)
void ln2_relu_out_kernel(const float* __restrict__ t, const float* __restrict__ g,
                         const float* __restrict__ bta, float* __restrict__ out,
                         int C, int N) {
  const int row = blockIdx.x * 8 + (threadIdx.x >> 5);   // b*N + n
  const int lane = threadIdx.x & 31;
  const float* p = t + (size_t)row * C;
  float s = 0.f, ss = 0.f;
  for (int c = lane; c < C; c += 32) { float v = p[c]; s += v; ss += v * v; }
#pragma unroll
  for (int o = 16; o; o >>= 1) { s += __shfl_xor(s, o, 32); ss += __shfl_xor(ss, o, 32); }
  const float mu = s / C;
  const float r = rsqrtf(ss / C - mu * mu + 1e-6f);
  const int b = row / N, n = row % N;
  for (int c = lane; c < C; c += 32) {
    float v = fmaxf((p[c] - mu) * r * g[c] + bta[c], 0.f);
    out[((size_t)b * C + c) * N + n] = v;
  }
}

// ---------------------------------------------------------------------------
extern "C" void kernel_launch(void* const* d_in, const int* in_sizes, int n_in,
                              void* d_out, int out_size, void* d_ws, size_t ws_size,
                              hipStream_t stream) {
  (void)in_sizes; (void)n_in; (void)out_size; (void)ws_size;
  const int B = 8, C = 512, N = 4096, F = 2048;

  const float* x    = (const float*)d_in[0];
  const float* w_qk = (const float*)d_in[1];
  const float* w_v  = (const float*)d_in[2];
  const float* w_phi= (const float*)d_in[3];
  const float* g1   = (const float*)d_in[4];
  const float* b1   = (const float*)d_in[5];
  const float* w1   = (const float*)d_in[6];
  const float* bb1  = (const float*)d_in[7];
  const float* w2   = (const float*)d_in[8];
  const float* bb2  = (const float*)d_in[9];
  const float* g2   = (const float*)d_in[10];
  const float* b2   = (const float*)d_in[11];
  float* out = (float*)d_out;

  // workspace carve (256B aligned)
  char* base = (char*)d_ws;
  size_t off = 0;
  auto carve = [&](size_t bytes) -> void* {
    off = (off + 255) & ~(size_t)255;
    void* p = base + off;
    off += bytes;
    return p;
  };
  u16*   wqkb = (u16*)carve((size_t)C * C * 2);
  u16*   wvb  = (u16*)carve((size_t)C * C * 2);
  u16*   wpb  = (u16*)carve((size_t)C * C * 2);
  u16*   w1b  = (u16*)carve((size_t)F * C * 2);
  u16*   w2b  = (u16*)carve((size_t)C * F * 2);
  u16*   xt   = (u16*)carve((size_t)B * N * C * 2);
  u16*   qc   = (u16*)carve((size_t)B * C * N * 2);
  u16*   qt   = (u16*)carve((size_t)B * N * C * 2);
  u16*   vc   = (u16*)carve((size_t)B * C * N * 2);
  u16*   kh   = (u16*)carve((size_t)B * C * N * 2);
  float* pphi = (float*)carve((size_t)B * C * N * 4);
  u16*   kvt  = (u16*)carve((size_t)B * C * C * 2);
  float* Qg   = (float*)carve((size_t)B * C * 4);
  float* sbuf = (float*)carve((size_t)B * N * 4);
  float* alph = (float*)carve((size_t)B * N * 4);
  float* ksum = (float*)carve((size_t)B * C * 4);
  float* zinv = (float*)carve((size_t)B * N * 4);
  float* uT   = (float*)carve((size_t)B * N * C * 4);
  u16*   hb   = (u16*)carve((size_t)B * N * C * 2);
  float* hf   = (float*)carve((size_t)B * N * C * 4);
  u16*   gT   = (u16*)carve((size_t)B * N * F * 2);
  float* tbuf = uT; // safe alias: uT consumed by ln1 before FFN2 writes

  // 1) weight conversion to bf16
  auto cvt = [&](const float* in, u16* o, size_t n) {
    cvt_f32_bf16_kernel<<<(unsigned)((n + 255) / 256), 256, 0, stream>>>(in, o, n);
  };
  cvt(w_qk, wqkb, (size_t)C * C);
  cvt(w_v,  wvb,  (size_t)C * C);
  cvt(w_phi,wpb,  (size_t)C * C);
  cvt(w1,   w1b,  (size_t)F * C);
  cvt(w2,   w2b,  (size_t)C * F);

  // 2) x -> xt (bf16, [B,N,C])
  transpose_cvt_kernel<<<dim3(N / 32, C / 32, B), 256, 0, stream>>>(x, xt, C, N);

  // 3) projections (WMMA GEMMs)
  dim3 gProj(N / BN, C / BM, B);
  gemm_qk_kernel <<<gProj, 256, 0, stream>>>(wqkb, xt, qc, qt, C, N);
  gemm_v_kernel  <<<gProj, 256, 0, stream>>>(wvb,  xt, vc, C, N);
  gemm_phi_kernel<<<gProj, 256, 0, stream>>>(wpb,  xt, pphi, C, N);

  // 4) alpha gating
  rowsum_bf16_kernel<<<(B * C) / 8, 256, 0, stream>>>(qc, Qg, N, 1.0f / (float)N);
  rowdot_kernel<<<(B * N) / 8, 256, 0, stream>>>(qt, Qg, sbuf, C, N, 0);
  softmax_scale_kernel<<<B, 1024, 0, stream>>>(sbuf, alph, N);
  {
    size_t total = (size_t)B * C * N;
    khat_kernel<<<(unsigned)((total + 255) / 256), 256, 0, stream>>>(qc, alph, kh, C, N, total);
  }
  rowsum_bf16_kernel<<<(B * C) / 8, 256, 0, stream>>>(kh, ksum, N, 1.0f);

  // 5) linear attention
  const float invSqrtC = 0.044194173824159216f; // 1/sqrt(512)
  gemm_kv_kernel<<<dim3(C / BN, C / BM, B), 256, 0, stream>>>(kh, vc, kvt, C, N, invSqrtC);
  rowdot_kernel<<<(B * N) / 8, 256, 0, stream>>>(qt, ksum, zinv, C, N, 1);
  gemm_y_kernel<<<dim3(N / BN, C / BM, B), 256, 0, stream>>>(kvt, qt, zinv, x, pphi, uT, C, N);

  // 6) LayerNorm 1
  ln1_kernel<<<(B * N) / 8, 256, 0, stream>>>(uT, g1, b1, hb, hf, C);

  // 7) FFN (WMMA GEMMs over B*N tokens)
  gemm_ffn1_kernel<<<dim3((B * N) / BN, F / BM, 1), 256, 0, stream>>>(w1b, hb, bb1, gT, C, F);
  gemm_ffn2_kernel<<<dim3((B * N) / BN, C / BM, 1), 256, 0, stream>>>(w2b, gT, bb2, hf, tbuf, F, C);

  // 8) LayerNorm 2 + relu + transpose to [B,C,N]
  ln2_relu_out_kernel<<<(B * N) / 8, 256, 0, stream>>>(tbuf, g2, b2, out, C, N);
}
